// RPolicy_90769838833988
// MI455X (gfx1250) — compile-verified, matrix operand order checked by reading, not run on hardware
//
#include <hip/hip_runtime.h>
#include <hip/hip_bf16.h>
#include <math.h>
#include <stdint.h>

// ---------------------------------------------------------------------------
// Types for CDNA5 WMMA (wave32): v_wmma_f32_16x16x32_bf16
// ---------------------------------------------------------------------------
typedef __bf16 bf16_t;
typedef bf16_t v16bf __attribute__((ext_vector_type(16)));
typedef float  v8f   __attribute__((ext_vector_type(8)));

static __device__ __forceinline__ uint16_t f32_to_bf16_bits(float f) {
  union { float f; unsigned u; } x; x.f = f;
  unsigned r = x.u + 0x7FFFu + ((x.u >> 16) & 1u);   // round-to-nearest-even
  return (uint16_t)(r >> 16);
}

// 16-bit 16x32 (rows x K) operand fragment from a row-major bf16 array
// (row = M for A, row = N for our transposed-B). ISA layout: lanes 0-15 ->
// rows, K pairs {0..7,16..23}; lanes 16-31 -> same rows, K +8.
static __device__ __forceinline__ v16bf
load_frag_rm(const uint16_t* __restrict__ base, int ld, int row0, int k0, int lane) {
  int r  = row0 + (lane & 15);
  int kb = k0 + ((lane >> 4) << 3);
  const uint16_t* p = base + (size_t)r * ld + kb;
  union { unsigned u[8]; v16bf v; } f;
#pragma unroll
  for (int i = 0; i < 4; ++i) {
    f.u[i]     = *(const unsigned*)(p + 2 * i);
    f.u[i + 4] = *(const unsigned*)(p + 16 + 2 * i);
  }
  return f.v;
}

// B fragment with all 16 columns equal to vector xbf (mat-vec via WMMA).
static __device__ __forceinline__ v16bf
load_frag_vec(const uint16_t* xbf, int k0, int lane) {
  int kb = k0 + ((lane >> 4) << 3);
  const uint16_t* p = xbf + kb;
  union { unsigned u[8]; v16bf v; } f;
#pragma unroll
  for (int i = 0; i < 4; ++i) {
    f.u[i]     = *(const unsigned*)(p + 2 * i);     // LDS broadcast reads
    f.u[i + 4] = *(const unsigned*)(p + 16 + 2 * i);
  }
  return f.v;
}

static __device__ __forceinline__ v8f wmma_bf16(v16bf a, v16bf b, v8f c) {
  return __builtin_amdgcn_wmma_f32_16x16x32_bf16(false, a, false, b,
                                                 (short)0, c, false, false);
}

// ---------------------------------------------------------------------------
// Conversion kernels
// ---------------------------------------------------------------------------
__global__ void convert_f32_bf16(const float* __restrict__ src,
                                 uint16_t* __restrict__ dst, int n) {
  for (int i = blockIdx.x * blockDim.x + threadIdx.x; i < n;
       i += gridDim.x * blockDim.x)
    dst[i] = f32_to_bf16_bits(src[i]);
}

// dst[n*K + k] = bf16(src[(rowoff + k) * srcld + n])   (transpose + convert)
__global__ void transpose_convert(const float* __restrict__ src,
                                  uint16_t* __restrict__ dst,
                                  int N, int K, int srcld, int rowoff) {
  int idx = blockIdx.x * blockDim.x + threadIdx.x;
  if (idx >= N * K) return;
  int n = idx / K, k = idx - n * K;
  dst[idx] = f32_to_bf16_bits(src[(size_t)(rowoff + k) * srcld + n]);
}

// ---------------------------------------------------------------------------
// WMMA GEMM: C[M,N] = epilogue(A[M,K](bf16 row-major) * Bt[N,K](bf16) + bias)
// block = 256 thr (8 waves). Block tile 32x256; each wave 16x64 (1 A frag
// feeds 4 wmma). Double-buffered K loop. Grid = (M/32, N/256).
// EPI 0: relu -> bf16 out ; EPI 1: bias only -> f32 out
// ---------------------------------------------------------------------------
template <int EPI, int K>
__global__ __launch_bounds__(256) void gemm_bf16_kernel(
    const uint16_t* __restrict__ A, const uint16_t* __restrict__ Bt,
    const float* __restrict__ bias, void* __restrict__ Cout, int N) {
  const int lane = threadIdx.x & 31, wave = threadIdx.x >> 5;
  const int m0 = blockIdx.x * 32 + (wave >> 2) * 16;
  const int n0 = blockIdx.y * 256 + (wave & 3) * 64;

  v8f c0 = {}, c1 = {}, c2 = {}, c3 = {};
  v16bf a = load_frag_rm(A, K, m0, 0, lane);
  v16bf b0 = load_frag_rm(Bt, K, n0,      0, lane);
  v16bf b1 = load_frag_rm(Bt, K, n0 + 16, 0, lane);
  v16bf b2 = load_frag_rm(Bt, K, n0 + 32, 0, lane);
  v16bf b3 = load_frag_rm(Bt, K, n0 + 48, 0, lane);
#pragma unroll 2
  for (int k0 = 32; k0 < K; k0 += 32) {
    v16bf an  = load_frag_rm(A, K, m0, k0, lane);
    v16bf b0n = load_frag_rm(Bt, K, n0,      k0, lane);
    v16bf b1n = load_frag_rm(Bt, K, n0 + 16, k0, lane);
    v16bf b2n = load_frag_rm(Bt, K, n0 + 32, k0, lane);
    v16bf b3n = load_frag_rm(Bt, K, n0 + 48, k0, lane);
    c0 = wmma_bf16(a, b0, c0);
    c1 = wmma_bf16(a, b1, c1);
    c2 = wmma_bf16(a, b2, c2);
    c3 = wmma_bf16(a, b3, c3);
    a = an; b0 = b0n; b1 = b1n; b2 = b2n; b3 = b3n;
  }
  c0 = wmma_bf16(a, b0, c0);
  c1 = wmma_bf16(a, b1, c1);
  c2 = wmma_bf16(a, b2, c2);
  c3 = wmma_bf16(a, b3, c3);

  const int rbase = m0 + ((lane >> 4) << 3);
  v8f accs[4] = {c0, c1, c2, c3};
#pragma unroll
  for (int j = 0; j < 4; ++j) {
    int col = n0 + 16 * j + (lane & 15);
    float bb = bias[col];
#pragma unroll
    for (int i = 0; i < 8; ++i) {
      float v = accs[j][i] + bb;
      size_t idx = (size_t)(rbase + i) * N + col;
      if (EPI == 0) {
        v = fmaxf(v, 0.0f);
        ((uint16_t*)Cout)[idx] = f32_to_bf16_bits(v);
      } else {
        ((float*)Cout)[idx] = v;
      }
    }
  }
}

// ---------------------------------------------------------------------------
// Persistent GRU scan (1 block x 512 threads = 16 waves).
// Mat-vec out[j] = sum_k Wt[j][k]*x[k] on WMMA: A = 16 weight rows,
// B = x broadcast into all 16 columns; lanes 0/16 scatter valid rows to LDS.
// Layer-1 gate weights live in LDS (padded stride 258 vs 16-way conflicts);
// everything else streams from L2 with double-buffered fragments.
// ---------------------------------------------------------------------------
template <int K>
static __device__ __forceinline__ void
matvec_wmma(const uint16_t* __restrict__ Wt, int ldw, const uint16_t* xbf,
            float* outacc, int ncols, int wave, int lane) {
  const int ntiles = ncols >> 4;
  for (int tIdx = wave; tIdx < ntiles; tIdx += 16) {
    const int j0 = tIdx << 4;
    v8f c = {};
    v16bf a = load_frag_rm(Wt, ldw, j0, 0, lane);
    v16bf b = load_frag_vec(xbf, 0, lane);
#pragma unroll 4
    for (int k0 = 32; k0 < K; k0 += 32) {
      v16bf an = load_frag_rm(Wt, ldw, j0, k0, lane);
      v16bf bn = load_frag_vec(xbf, k0, lane);
      c = wmma_bf16(a, b, c);
      a = an; b = bn;
    }
    c = wmma_bf16(a, b, c);
    if ((lane & 15) == 0) {            // lane 0 -> rows j0..+7, lane16 -> +8..15
      int rbase = j0 + ((lane >> 4) << 3);
#pragma unroll
      for (int i = 0; i < 8; ++i) outacc[rbase + i] = c[i];
    }
  }
}

#define WHG1_LD        258                         // padded LDS row stride
#define WHG1_LDS_BYTES (512 * WHG1_LD * 2)         // 264192
#define SCAN_F32_BYTES (2560 * 4)                  // state floats
#define SCAN_U16_BYTES (1280 * 2)                  // packed bf16 vectors
#define SCAN_SMEM      (WHG1_LDS_BYTES + SCAN_F32_BYTES + SCAN_U16_BYTES)

__global__ __launch_bounds__(512) void scan_kernel(
    const float* __restrict__ Xg1, const float* __restrict__ Xc1,
    const uint16_t* __restrict__ Whg1t, const uint16_t* __restrict__ Whc1t,
    const uint16_t* __restrict__ W2gt,  const uint16_t* __restrict__ W2ct,
    const float* __restrict__ gb2, const float* __restrict__ cb2,
    float* __restrict__ ys, int T) {
  extern __shared__ __align__(16) char smem[];
  uint16_t* WhL  = (uint16_t*)smem;                        // layer1 gate W, LDS
  float*    fb   = (float*)(smem + WHG1_LDS_BYTES);
  float* h1f  = fb;        float* h2f   = fb + 256;
  float* u1   = fb + 512;  float* u2    = fb + 768;
  float* gacc = fb + 1024; float* cacc  = fb + 1536;
  float* g2acc= fb + 1792; float* c2acc = fb + 2304;
  uint16_t* ub = (uint16_t*)(smem + WHG1_LDS_BYTES + SCAN_F32_BYTES);
  uint16_t* cat2bf  = ub;          // [h1 | h2]     (bf16)
  uint16_t* catc2bf = ub + 512;    // [h1 | r2*h2]  (bf16)
  uint16_t* r1hbf   = ub + 1024;   // r1*h1         (bf16)

  const int tid = threadIdx.x, lane = tid & 31, wave = tid >> 5;
  if (tid < 256) { h1f[tid] = 0.f; h2f[tid] = 0.f; r1hbf[tid] = 0; }
  cat2bf[tid] = 0; catc2bf[tid] = 0;
  // stash layer-1 gate weights into LDS (one time, padded rows)
  for (int idx = tid; idx < 512 * 256; idx += 512) {
    int r = idx >> 8, k = idx & 255;
    WhL[r * WHG1_LD + k] = Whg1t[idx];
  }
  const float gb2r = gb2[tid];
  const float cb2r = (tid < 256) ? cb2[tid] : 0.f;
  __syncthreads();

  for (int t = 0; t < T; ++t) {
    float xg = Xg1[(size_t)t * 512 + tid];
    float xc = (tid < 256) ? Xc1[(size_t)t * 256 + tid] : 0.f;
    if (t + 1 < T) {                                 // gfx1250 global_prefetch_b8
      __builtin_prefetch(&Xg1[(size_t)(t + 1) * 512 + tid], 0, 0);
      if (tid < 256) __builtin_prefetch(&Xc1[(size_t)(t + 1) * 256 + tid], 0, 0);
    }
    // --- layer1 gates: gi1 = Xg1[t] + h1 @ Whg1   (weights from LDS) ---
    matvec_wmma<256>(WhL, WHG1_LD, cat2bf, gacc, 512, wave, lane);
    __syncthreads();
    {
      float g = 1.f / (1.f + __expf(-(gacc[tid] + xg)));
      if (tid < 256) r1hbf[tid] = f32_to_bf16_bits(g * h1f[tid]);  // r1*h1
      else           u1[tid - 256] = g;
    }
    __syncthreads();
    // --- layer1 candidate: c1 = tanh(Xc1[t] + (r1*h1) @ Whc1) ---
    matvec_wmma<256>(Whc1t, 256, r1hbf, cacc, 256, wave, lane);
    __syncthreads();
    if (tid < 256) {
      float cc = tanhf(cacc[tid] + xc);
      float u  = u1[tid];
      float h  = u * h1f[tid] + (1.f - u) * cc;
      h1f[tid] = h;
      uint16_t hb = f32_to_bf16_bits(h);
      cat2bf[tid] = hb; catc2bf[tid] = hb;
    }
    __syncthreads();
    // --- layer2 gates: gi2 = gb2 + [h1,h2] @ W2g ---
    matvec_wmma<512>(W2gt, 512, cat2bf, g2acc, 512, wave, lane);
    __syncthreads();
    {
      float g = 1.f / (1.f + __expf(-(g2acc[tid] + gb2r)));
      if (tid < 256) catc2bf[256 + tid] = f32_to_bf16_bits(g * h2f[tid]);
      else           u2[tid - 256] = g;
    }
    __syncthreads();
    // --- layer2 candidate: c2 = tanh(cb2 + [h1, r2*h2] @ W2c) ---
    matvec_wmma<512>(W2ct, 512, catc2bf, c2acc, 256, wave, lane);
    __syncthreads();
    if (tid < 256) {
      float cc = tanhf(c2acc[tid] + cb2r);
      float u  = u2[tid];
      float h  = u * h2f[tid] + (1.f - u) * cc;
      h2f[tid] = h;
      cat2bf[256 + tid] = f32_to_bf16_bits(h);
      ys[(size_t)t * 256 + tid] = h;
    }
    __syncthreads();
  }
}

// ---------------------------------------------------------------------------
// Output heads (tiny, bandwidth bound): logits[t][18], value[t]
// ---------------------------------------------------------------------------
__global__ __launch_bounds__(32) void heads_kernel(
    const float* __restrict__ ys, const float* __restrict__ pol_w,
    const float* __restrict__ pol_b, const float* __restrict__ val_w,
    const float* __restrict__ val_b, float* __restrict__ logits,
    float* __restrict__ value) {
  int t = blockIdx.x, a = threadIdx.x;
  const float* y = ys + (size_t)t * 256;
  if (a < 18) {
    float s = pol_b[a];
    for (int k = 0; k < 256; ++k) s += y[k] * pol_w[k * 18 + a];
    logits[(size_t)t * 18 + a] = s;
  } else if (a == 18) {
    float s = val_b[0];
    for (int k = 0; k < 256; ++k) s += y[k] * val_w[k];
    value[t] = s;
  }
}

// ---------------------------------------------------------------------------
extern "C" void kernel_launch(void* const* d_in, const int* in_sizes, int n_in,
                              void* d_out, int out_size, void* d_ws, size_t ws_size,
                              hipStream_t stream) {
  (void)in_sizes; (void)n_in; (void)out_size; (void)ws_size;
  const int T = 16384, D = 512, H = 256;

  const float* obs   = (const float*)d_in[0];
  const float* fc1_w = (const float*)d_in[1];
  const float* fc1_b = (const float*)d_in[2];
  const float* g1_gk = (const float*)d_in[3];
  const float* g1_gb = (const float*)d_in[4];
  const float* g1_ck = (const float*)d_in[5];
  const float* g1_cb = (const float*)d_in[6];
  const float* g2_gk = (const float*)d_in[7];
  const float* g2_gb = (const float*)d_in[8];
  const float* g2_ck = (const float*)d_in[9];
  const float* g2_cb = (const float*)d_in[10];
  const float* pol_w = (const float*)d_in[11];
  const float* pol_b = (const float*)d_in[12];
  const float* val_w = (const float*)d_in[13];
  const float* val_b = (const float*)d_in[14];

  char* ws = (char*)d_ws;
  size_t off = 0;
  auto alloc = [&](size_t bytes) -> void* {
    off = (off + 255) & ~(size_t)255;
    void* p = ws + off;
    off += bytes;
    return p;
  };
  uint16_t* X0    = (uint16_t*)alloc((size_t)T * D * 2);     // obs bf16
  uint16_t* Wfc1t = (uint16_t*)alloc((size_t)H * D * 2);     // [256][512]
  uint16_t* Wxg1t = (uint16_t*)alloc((size_t)512 * 256 * 2); // g1_gk input rows^T
  uint16_t* Wxc1t = (uint16_t*)alloc((size_t)256 * 256 * 2);
  uint16_t* Whg1t = (uint16_t*)alloc((size_t)512 * 256 * 2); // g1_gk recurrent^T
  uint16_t* Whc1t = (uint16_t*)alloc((size_t)256 * 256 * 2);
  uint16_t* W2gt  = (uint16_t*)alloc((size_t)512 * 512 * 2);
  uint16_t* W2ct  = (uint16_t*)alloc((size_t)256 * 512 * 2);
  uint16_t* X     = (uint16_t*)alloc((size_t)T * H * 2);     // fc1 out bf16
  float*    Xg1   = (float*)alloc((size_t)T * 512 * 4);
  float*    Xc1   = (float*)alloc((size_t)T * 256 * 4);
  float*    ysbuf = (float*)alloc((size_t)T * 256 * 4);

  // 1) convert obs + weights (transposed to output-major x K for fragments)
  convert_f32_bf16<<<4096, 256, 0, stream>>>(obs, X0, T * D);
  transpose_convert<<<(256 * 512 + 255) / 256, 256, 0, stream>>>(fc1_w, Wfc1t, 256, 512, 256, 0);
  transpose_convert<<<(512 * 256 + 255) / 256, 256, 0, stream>>>(g1_gk, Wxg1t, 512, 256, 512, 0);
  transpose_convert<<<(512 * 256 + 255) / 256, 256, 0, stream>>>(g1_gk, Whg1t, 512, 256, 512, 256);
  transpose_convert<<<(256 * 256 + 255) / 256, 256, 0, stream>>>(g1_ck, Wxc1t, 256, 256, 256, 0);
  transpose_convert<<<(256 * 256 + 255) / 256, 256, 0, stream>>>(g1_ck, Whc1t, 256, 256, 256, 256);
  transpose_convert<<<(512 * 512 + 255) / 256, 256, 0, stream>>>(g2_gk, W2gt, 512, 512, 512, 0);
  transpose_convert<<<(256 * 512 + 255) / 256, 256, 0, stream>>>(g2_ck, W2ct, 256, 512, 256, 0);

  // 2) fc1: X = relu(obs @ fc1_w + b)   [T,256] bf16
  gemm_bf16_kernel<0, 512><<<dim3(T / 32, 1), 256, 0, stream>>>(X0, Wfc1t, fc1_b, X, 256);
  // 3) precompute input-dependent GRU1 contributions (parallel GEMMs)
  gemm_bf16_kernel<1, 256><<<dim3(T / 32, 2), 256, 0, stream>>>(X, Wxg1t, g1_gb, Xg1, 512);
  gemm_bf16_kernel<1, 256><<<dim3(T / 32, 1), 256, 0, stream>>>(X, Wxc1t, g1_cb, Xc1, 256);
  // 4) sequential scan on one persistent WGP, mat-vecs on WMMA,
  //    layer-1 gate weights LDS-resident (dynamic LDS, 320KB WGP budget)
  scan_kernel<<<1, 512, SCAN_SMEM, stream>>>(Xg1, Xc1, Whg1t, Whc1t, W2gt, W2ct,
                                             g2_gb, g2_cb, ysbuf, T);
  // 5) heads
  float* logits = (float*)d_out;
  float* value  = logits + (size_t)T * 18;
  heads_kernel<<<T, 32, 0, stream>>>(ysbuf, pol_w, pol_b, val_w, val_b, logits, value);
}
